// CustomRNN_24438363914844
// MI455X (gfx1250) — matrix operand order repaired and checked by reading, not compiled, not run on hardware
//
#include <hip/hip_runtime.h>

typedef __attribute__((ext_vector_type(16))) _Float16 v16h;
typedef __attribute__((ext_vector_type(8)))  _Float16 v8h;
typedef __attribute__((ext_vector_type(8)))  float    v8f;
typedef __attribute__((ext_vector_type(4)))  float    v4f;

#define RNN_CLIP 10.0f

// LDS tile: 128 columns of R^T, K=512, padded to kill bank conflicts
#define LDS_COLS   128
#define LDS_STRIDE 520   // 512 + 8 halves pad -> 1040B row stride (16B aligned)
#define LDS_BYTES  (LDS_COLS * LDS_STRIDE * 2)

// ---------------------------------------------------------------------------
// Prep kernels
// ---------------------------------------------------------------------------
// fp32 row-major [rows x cols] -> fp16 transposed [cols x rows]
__global__ void transpose_cvt_f32_f16(const float* __restrict__ src,
                                      _Float16* __restrict__ dst,
                                      int rows, int cols) {
  int idx = blockIdx.x * blockDim.x + threadIdx.x;
  if (idx >= rows * cols) return;
  int r = idx / cols, c = idx % cols;        // coalesced read of src
  dst[c * rows + r] = (_Float16)src[r * cols + c];
}

__global__ void cvt_f32_f16(const float* __restrict__ src,
                            _Float16* __restrict__ dst, int n) {
  int i = blockIdx.x * blockDim.x + threadIdx.x;
  if (i < n) dst[i] = (_Float16)src[i];
}

__global__ void zero_u32(unsigned* __restrict__ p, int n) {
  int i = blockIdx.x * blockDim.x + threadIdx.x;
  if (i < n) p[i] = 0u;
}

// ---------------------------------------------------------------------------
// WMMA fragment loaders (wave32, 16x16x32 f16 shapes per CDNA5 ISA 7.12.2)
// A (16x32):
//   lane<16: row=lane,    K chunks [kBase+0..7]  and [kBase+16..23]
//   lane>=16:row=lane-16, K chunks [kBase+8..15] and [kBase+24..31]
// ---------------------------------------------------------------------------
// A from fp32 row-major source (convert in-register) — used by the big GEMM
__device__ __forceinline__ v16h load_A_f32(const float* __restrict__ base, int ld,
                                           int rowBase, int kBase, int lane) {
  int m  = rowBase + (lane & 15);
  int kc = kBase + ((lane < 16) ? 0 : 8);
  const float* p = base + (size_t)m * ld + kc;
  v4f f0 = *(const v4f*)(p);
  v4f f1 = *(const v4f*)(p + 4);
  v4f f2 = *(const v4f*)(p + 16);
  v4f f3 = *(const v4f*)(p + 20);
  v16h a;
#pragma unroll
  for (int j = 0; j < 4; ++j) {
    a[j]      = (_Float16)f0[j];
    a[4 + j]  = (_Float16)f1[j];
    a[8 + j]  = (_Float16)f2[j];
    a[12 + j] = (_Float16)f3[j];
  }
  return a;
}

// A from f16 row-major source (no conversion) — used by the recurrent scan
__device__ __forceinline__ v16h load_A_f16(const _Float16* __restrict__ base,
                                           int ld, int rowBase, int kBase,
                                           int lane) {
  int m  = rowBase + (lane & 15);
  int kc = kBase + ((lane < 16) ? 0 : 8);
  const _Float16* p = base + (size_t)m * ld + kc;  // 16B aligned
  v8h lo = *(const v8h*)(p);                       // K = kc .. kc+7
  v8h hi = *(const v8h*)(p + 16);                  // K = kc+16 .. kc+23
  v16h a;
#pragma unroll
  for (int j = 0; j < 8; ++j) { a[j] = lo[j]; a[8 + j] = hi[j]; }
  return a;
}

// B (32x16) from pre-transposed f16 [N x K] row-major in GLOBAL memory:
__device__ __forceinline__ v16h load_Bt(const _Float16* __restrict__ bt, int ldk,
                                        int nBase, int kBase, int lane) {
  int n = nBase + (lane & 15);
  int k = kBase + ((lane < 16) ? 0 : 8);
  const _Float16* p = bt + (size_t)n * ldk + k;      // 16B aligned -> 2x b128
  v8h lo = *(const v8h*)(p);
  v8h hi = *(const v8h*)(p + 16);
  v16h b;
#pragma unroll
  for (int j = 0; j < 8; ++j) { b[j] = lo[j]; b[8 + j] = hi[j]; }
  return b;
}

// B (32x16) from LDS-resident R^T slice (n relative to slice, padded stride):
__device__ __forceinline__ v16h load_Bt_lds(const _Float16* lds,
                                            int nRel, int kBase, int lane) {
  int n = nRel + (lane & 15);
  int k = kBase + ((lane < 16) ? 0 : 8);
  const _Float16* p = lds + n * LDS_STRIDE + k;      // 16B aligned
  v8h lo = *(const v8h*)(p);
  v8h hi = *(const v8h*)(p + 16);
  v16h b;
#pragma unroll
  for (int j = 0; j < 8; ++j) { b[j] = lo[j]; b[8 + j] = hi[j]; }
  return b;
}

// ---------------------------------------------------------------------------
// Kernel 2: G = X @ W + bias   (M = B*T rows), stored fp32 into d_out outputs
// ---------------------------------------------------------------------------
__global__ void __launch_bounds__(256)
gemm_xw_bias(const float* __restrict__ X,      // [M x K] fp32
             const _Float16* __restrict__ Wt,  // [H x K] f16 (transposed W)
             const float* __restrict__ bias,   // [H]
             float* __restrict__ G,            // [M x H] fp32
             int K, int H) {
  int lane = threadIdx.x & 31;
  int wave = threadIdx.x >> 5;
  int gwid = blockIdx.x * 8 + wave;
  int mtile  = gwid >> 3;
  int ngroup = gwid & 7;
  int rowBase = mtile * 16;
  int nBase   = ngroup * 64;

  v8f acc[4];
#pragma unroll
  for (int j = 0; j < 4; ++j) {
    float bv = bias[nBase + j * 16 + (lane & 15)];   // column fixed per lane
#pragma unroll
    for (int v = 0; v < 8; ++v) acc[j][v] = bv;
  }

  for (int k = 0; k < K; k += 32) {
    v16h a = load_A_f32(X, K, rowBase, k, lane);
#pragma unroll
    for (int j = 0; j < 4; ++j) {
      v16h b = load_Bt(Wt, K, nBase + j * 16, k, lane);
      acc[j] = __builtin_amdgcn_wmma_f32_16x16x32_f16(
          false, a, false, b, (short)0, acc[j], false, false);
    }
  }

  int mloc = 8 * (lane >> 4);
#pragma unroll
  for (int j = 0; j < 4; ++j) {
    int col = nBase + j * 16 + (lane & 15);
#pragma unroll
    for (int v = 0; v < 8; ++v)
      G[(size_t)(rowBase + mloc + v) * H + col] = acc[j][v];
  }
}

// ---------------------------------------------------------------------------
// Kernel 3: persistent recurrent scan. 16 blocks x 8 waves = 128 wave-tiles
//   block b: mtile = b>>2 (16 batch rows), ntiles (b&3)*8 .. +7 (128 cols)
// R^T slice staged in LDS once, reused for all T steps. h ping-pong buffers
// are f16 (WMMA consumes f16 anyway; fp32 h goes to d_out). Steps separated
// by a device-scope arrival-counter barrier (counters zeroed every call).
// ---------------------------------------------------------------------------
__global__ void __launch_bounds__(256)
rnn_scan(const _Float16* __restrict__ hinit, // [B x H] f16 (converted h0)
         const _Float16* __restrict__ Rt,    // [H x H] f16 (transposed R)
         float* __restrict__ out,            // [B x T x H] fp32 (holds G)
         _Float16* __restrict__ hbuf0,       // [B x H] f16 ping
         _Float16* __restrict__ hbuf1,       // [B x H] f16 pong
         float* __restrict__ hlast,          // [B x H] fp32 tail of d_out
         unsigned* __restrict__ ctr,         // [T] arrival counters (zeroed)
         int T, int H, int numBlocks) {
  extern __shared__ _Float16 ldsB[];

  int lane = threadIdx.x & 31;
  int wave = threadIdx.x >> 5;
  int blk  = blockIdx.x;
  int mtile   = blk >> 2;
  int colBase = (blk & 3) * LDS_COLS;
  int ntile   = (blk & 3) * 8 + wave;
  int rowBase = mtile * 16;
  int nBase   = ntile * 16;
  int nRel    = nBase - colBase;             // column offset inside LDS slice

  // ---- Stage R^T[colBase .. colBase+127][0..H) into LDS (once) ----
  {
    const int chunks = LDS_COLS * (512 / 8);               // 8-half chunks
    for (int i = threadIdx.x; i < chunks; i += 256) {
      int n    = i >> 6;                                    // 64 chunks per row
      int koff = (i & 63) * 8;
      *(v8h*)(ldsB + n * LDS_STRIDE + koff) =
          *(const v8h*)(Rt + (size_t)(colBase + n) * H + koff);
    }
  }
  __syncthreads();

  for (int t = 0; t < T; ++t) {
    const _Float16* hp = (t == 0) ? hinit : ((t & 1) ? hbuf0 : hbuf1);
    _Float16*       hn = (t & 1) ? hbuf1 : hbuf0;

    v8f acc;
#pragma unroll
    for (int v = 0; v < 8; ++v) acc[v] = 0.0f;

#pragma unroll 4
    for (int k = 0; k < H; k += 32) {
      v16h a = load_A_f16(hp, H, rowBase, k, lane);
      v16h b = load_Bt_lds(ldsB, nRel, k, lane);
      acc = __builtin_amdgcn_wmma_f32_16x16x32_f16(
          false, a, false, b, (short)0, acc, false, false);
    }

    int col  = nBase + (lane & 15);
    int mloc = 8 * (lane >> 4);
#pragma unroll
    for (int v = 0; v < 8; ++v) {
      int b = rowBase + mloc + v;
      size_t gi = ((size_t)b * T + t) * H + col;
      float g = out[gi] + acc[v];              // G already holds x@W + bias
      g = fminf(fmaxf(g, -RNN_CLIP), RNN_CLIP);
      float h = tanhf(g);
      out[gi] = h;                             // overwrite G with h_t (fp32)
      hn[(size_t)b * H + col] = (_Float16)h;   // f16 state for next step's A
      if (t == T - 1) hlast[(size_t)b * H + col] = h;
    }

    // ---- device-scope step barrier (fresh counter per step) ----
    __threadfence();                           // my h writes visible agent-wide
    __syncthreads();                           // whole block done
    if (threadIdx.x == 0) {
      __hip_atomic_fetch_add(&ctr[t], 1u, __ATOMIC_RELEASE,
                             __HIP_MEMORY_SCOPE_AGENT);
      while (__hip_atomic_load(&ctr[t], __ATOMIC_ACQUIRE,
                               __HIP_MEMORY_SCOPE_AGENT) < (unsigned)numBlocks)
        __builtin_amdgcn_s_sleep(1);
    }
    __syncthreads();
    __threadfence();                           // acquire: discard stale h lines
  }
}

// ---------------------------------------------------------------------------
extern "C" void kernel_launch(void* const* d_in, const int* in_sizes, int n_in,
                              void* d_out, int out_size, void* d_ws, size_t ws_size,
                              hipStream_t stream) {
  const float* x    = (const float*)d_in[0];   // [B,T,I]
  const float* W    = (const float*)d_in[1];   // [I,H]
  const float* R    = (const float*)d_in[2];   // [H,H]
  const float* bias = (const float*)d_in[3];   // [H]
  const float* h0   = (const float*)d_in[4];   // [B,H]

  const int Bsz = 64, T = 512, I = 512, H = 512;
  const int M = Bsz * T;
  const int NBLK = 16;

  char* ws = (char*)d_ws;
  _Float16* Wt    = (_Float16*)ws;                                  // H*I f16
  _Float16* Rt    = Wt + (size_t)H * I;                             // H*H f16
  _Float16* hinit = Rt + (size_t)H * H;                             // B*H f16
  _Float16* hbuf0 = hinit + (size_t)Bsz * H;                        // B*H f16
  _Float16* hbuf1 = hbuf0 + (size_t)Bsz * H;                        // B*H f16
  unsigned* ctr   = (unsigned*)(hbuf1 + (size_t)Bsz * H);           // T u32

  float* outp  = (float*)d_out;               // [B,T,H]
  float* hlast = outp + (size_t)M * H;        // [B,H]

  // deterministic per-call init of barrier counters + f16 initial state
  zero_u32<<<(T + 255) / 256, 256, 0, stream>>>(ctr, T);
  cvt_f32_f16<<<(Bsz * H + 255) / 256, 256, 0, stream>>>(h0, hinit, Bsz * H);

  int n1 = I * H;
  transpose_cvt_f32_f16<<<(n1 + 255) / 256, 256, 0, stream>>>(W, Wt, I, H);
  transpose_cvt_f32_f16<<<(n1 + 255) / 256, 256, 0, stream>>>(R, Rt, H, H);

  // G = X@W + bias -> d_out outputs region. 2048 blocks x 8 waves.
  gemm_xw_bias<<<M / 16, 256, 0, stream>>>(x, Wt, bias, outp, I, H);

  // Single persistent kernel runs the whole T-step scan.
  rnn_scan<<<NBLK, 256, LDS_BYTES, stream>>>(hinit, Rt, outp, hbuf0, hbuf1,
                                             hlast, ctr, T, H, NBLK);
}
